// EncoderBlock_51084341018719
// MI455X (gfx1250) — matrix operand order repaired and checked by reading, not compiled, not run on hardware
//
#include <hip/hip_runtime.h>
#include <math.h>

#define B_     24
#define C_     128
#define H_     8
#define DK_    16
#define L_     512
#define KW     7
#define NCONV_ 4

typedef __attribute__((ext_vector_type(16))) _Float16 v16h;
typedef __attribute__((ext_vector_type(8)))  _Float16 v8h;
typedef __attribute__((ext_vector_type(8)))  float    v8f;

// ---------------------------------------------------------------------------
// WMMA helpers (CDNA5 16x16x32 f16 -> f32)
// A layout (16x32 f16, row-major source, lda in halves):
//   lane<16 : row=lane,    K = {0..7, 16..23}
//   lane>=16: row=lane-16, K = {8..15, 24..31}
// B layout (32x16 f16, N-major source Bn[n][k], ldb in halves):
//   lane<16 : col=lane,    K = {0..7, 16..23}
//   lane>=16: col=lane-16, K = {8..15, 24..31}
// D layout (16x16 f32): reg r -> (M = r + (lane<16?0:8), N = lane%16)
// ---------------------------------------------------------------------------
__device__ inline v16h load_a16(const _Float16* A, int lda) {
    int lane = threadIdx.x & 31;
    int row  = lane & 15;
    int k0   = (lane < 16) ? 0 : 8;
    const _Float16* p = A + row * lda + k0;
    union { v16h v; v8h h[2]; } u;
    u.h[0] = *(const v8h*)(p);
    u.h[1] = *(const v8h*)(p + 16);
    return u.v;
}

__device__ inline v16h load_b16(const _Float16* Bn, int ldb) {
    int lane = threadIdx.x & 31;
    int col  = lane & 15;
    int k0   = (lane < 16) ? 0 : 8;
    const _Float16* p = Bn + col * ldb + k0;
    union { v16h v; v8h h[2]; } u;
    u.h[0] = *(const v8h*)(p);
    u.h[1] = *(const v8h*)(p + 16);
    return u.v;
}

__device__ inline v8f wmma_f16(v16h a, v16h b, v8f c) {
    return __builtin_amdgcn_wmma_f32_16x16x32_f16(
        /*neg_a=*/false, a, /*neg_b=*/false, b,
        /*c_mod=*/(short)0, c, /*reuse_a=*/false, /*reuse_b=*/false);
}

// Deterministic dropout mask (p=0.1 keep-scale 1/0.9). Not bit-identical to
// JAX threefry, but deterministic per (seed, element) across graph replays.
__device__ inline float drop_scale(unsigned seed, unsigned idx) {
    unsigned h = idx * 2654435761u + seed * 0x9E3779B9u;
    h ^= h >> 16; h *= 0x7feb352du;
    h ^= h >> 15; h *= 0x846ca68bu;
    h ^= h >> 16;
    return (h >= 429496730u) ? (1.0f / 0.9f) : 0.0f;  // drop ~10%
}

// ---------------------------------------------------------------------------
// f32 -> f16 weight conversion (row-major A operands)
// ---------------------------------------------------------------------------
__global__ void k_cvt(const float* __restrict__ src, _Float16* __restrict__ dst, int n) {
    int i = blockIdx.x * 256 + threadIdx.x;
    if (i < n) dst[i] = (_Float16)src[i];
}

// ---------------------------------------------------------------------------
// Positional encoding: cur = x + sin(sin(pos*freq + phase))
// ---------------------------------------------------------------------------
__global__ void k_posenc(const float* __restrict__ x, float* __restrict__ cur) {
    int i = blockIdx.x * 256 + threadIdx.x;          // over B*C*L
    int l = i % L_;
    int c = (i / L_) % C_;
    int ce = c & ~1;
    float base  = __powf(10000.f, -(float)ce / (float)C_);
    float freq  = (c & 1) ? -base : base;
    float phase = (c & 1) ? 1.57079632679f : 0.f;
    float pe = __sinf(__sinf((float)l * freq + phase));
    cur[i] = x[i] + pe;
}

// ---------------------------------------------------------------------------
// Row norm over L (mean, unbiased std, eps=1e-6). One wave per (b,c) row.
// Optionally writes f32 row-major and/or f16 transposed (N-major B operand).
// ---------------------------------------------------------------------------
__global__ void k_norm(const float* __restrict__ in,
                       float* __restrict__ out_f32,
                       _Float16* __restrict__ outT16) {
    int wave = threadIdx.x >> 5, lane = threadIdx.x & 31;
    int row  = blockIdx.x * 8 + wave;                // [0, B*C)
    int b = row / C_, c = row % C_;
    const float* p = in + (size_t)row * L_;
    float vals[16];
    float s = 0.f;
#pragma unroll
    for (int i = 0; i < 16; i++) { vals[i] = p[lane + 32 * i]; s += vals[i]; }
#pragma unroll
    for (int o = 16; o > 0; o >>= 1) s += __shfl_xor(s, o);
    float mean = s / (float)L_;
    float ss = 0.f;
#pragma unroll
    for (int i = 0; i < 16; i++) { float d = vals[i] - mean; ss += d * d; }
#pragma unroll
    for (int o = 16; o > 0; o >>= 1) ss += __shfl_xor(ss, o);
    float inv = 1.f / (sqrtf(ss / (float)(L_ - 1)) + 1e-6f);
#pragma unroll
    for (int i = 0; i < 16; i++) {
        float v = (vals[i] - mean) * inv;
        int l = lane + 32 * i;
        if (out_f32) out_f32[(size_t)row * L_ + l] = v;
        if (outT16)  outT16[((size_t)b * L_ + l) * C_ + c] = (_Float16)v;
    }
}

// ---------------------------------------------------------------------------
// Depthwise conv K=7, same padding, + bias. Emits f16 transposed (L,C) for GEMM.
// ---------------------------------------------------------------------------
__global__ void k_dwconv(const float* __restrict__ nrm,
                         const float* __restrict__ dw_w,
                         const float* __restrict__ dw_b,
                         int ci, _Float16* __restrict__ outT16) {
    int i = blockIdx.x * 256 + threadIdx.x;          // over B*C*L
    int l = i % L_;
    int c = (i / L_) % C_;
    int b = i / (L_ * C_);
    const float* w = dw_w + ((size_t)ci * C_ + c) * KW;
    const float* r = nrm + ((size_t)b * C_ + c) * L_;
    float acc = dw_b[ci * C_ + c];
#pragma unroll
    for (int t = 0; t < KW; t++) {
        int ll = l + t - KW / 2;
        if (ll >= 0 && ll < L_) acc += r[ll] * w[t];
    }
    outT16[((size_t)b * L_ + l) * C_ + c] = (_Float16)acc;
}

// ---------------------------------------------------------------------------
// Generic C x L GEMM (M=128, N=512, K=128 per batch) with fused epilogue:
//   v = A@B (+bias) ; (relu) ; out = (res + v) * dropout(seed)
// A16 row-major f16 (stride a_stride_b per batch; 0 = shared weight),
// Bn N-major f16 (B, L, C). One wave per 16x16 tile, 8 waves/WG.
// ---------------------------------------------------------------------------
__global__ void __launch_bounds__(256) k_gemm(
        const _Float16* __restrict__ A16, int a_stride_b,
        const _Float16* __restrict__ Bn,
        const float* __restrict__ bias,
        const float* __restrict__ res,
        float* __restrict__ out,
        int relu, unsigned seed) {
    int wave = threadIdx.x >> 5;
    int nt = blockIdx.x * 8 + wave;                  // 0..31
    int mt = blockIdx.y;                             // 0..7
    int b  = blockIdx.z;
    const _Float16* A  = A16 + (size_t)b * a_stride_b + mt * 16 * C_;
    const _Float16* Bb = Bn + ((size_t)b * L_ + nt * 16) * C_;
    v8f acc = {};
#pragma unroll
    for (int kt = 0; kt < C_ / 32; kt++) {
        v16h a  = load_a16(A + kt * 32, C_);
        v16h bt = load_b16(Bb + kt * 32, C_);
        acc = wmma_f16(a, bt, acc);
    }
    int lane = threadIdx.x & 31;
    int n  = nt * 16 + (lane & 15);
    int m0 = mt * 16 + ((lane < 16) ? 0 : 8);
#pragma unroll
    for (int r = 0; r < 8; r++) {
        int m = m0 + r;
        float v = acc[r];
        if (bias) v += bias[m];
        if (relu) v = fmaxf(v, 0.f);
        unsigned idx = ((unsigned)(b * C_ + m)) * L_ + n;
        out[idx] = (res[idx] + v) * drop_scale(seed, idx);
    }
}

// ---------------------------------------------------------------------------
// QKV projection: per (h,b): {q,k,v}(16,512) = W{q,k,v}(16,128) @ xn(128,512)
//   q,k stored transposed+padded [hb][L][32] (rows: A for scores, B N-major),
//   v stored row-major [hb][16][512] (natural N-major B for the head GEMM).
// ---------------------------------------------------------------------------
__global__ void __launch_bounds__(256) k_qkv(
        const _Float16* __restrict__ Wq16, const _Float16* __restrict__ Wk16,
        const _Float16* __restrict__ Wv16, const _Float16* __restrict__ xnT,
        _Float16* __restrict__ qT, _Float16* __restrict__ kT,
        _Float16* __restrict__ Vm) {
    int wave = threadIdx.x >> 5, lane = threadIdx.x & 31;
    int nt = blockIdx.x * 8 + wave;                  // 0..31
    int hb = blockIdx.y;                             // h*B + b, 0..191
    int b  = hb % B_;
    const _Float16* Bb = xnT + ((size_t)b * L_ + nt * 16) * C_;
    const _Float16* Aq = Wq16 + (size_t)hb * DK_ * C_;
    const _Float16* Ak = Wk16 + (size_t)hb * DK_ * C_;
    const _Float16* Av = Wv16 + (size_t)hb * DK_ * C_;
    v8f aq = {}, ak = {}, av = {};
#pragma unroll
    for (int kt = 0; kt < C_ / 32; kt++) {
        v16h bt = load_b16(Bb + kt * 32, C_);
        aq = wmma_f16(load_a16(Aq + kt * 32, C_), bt, aq);
        ak = wmma_f16(load_a16(Ak + kt * 32, C_), bt, ak);
        av = wmma_f16(load_a16(Av + kt * 32, C_), bt, av);
    }
    int nl = lane & 15;
    int m0 = (lane < 16) ? 0 : 8;
    int l  = nt * 16 + nl;
    _Float16* qrow = qT + ((size_t)hb * L_ + l) * 32;
    _Float16* krow = kT + ((size_t)hb * L_ + l) * 32;
#pragma unroll
    for (int r = 0; r < 8; r++) {
        qrow[m0 + r]      = (_Float16)aq[r];
        krow[m0 + r]      = (_Float16)ak[r];
        qrow[m0 + 16 + r] = (_Float16)0.f;   // zero K-padding 16..31
        krow[m0 + 16 + r] = (_Float16)0.f;
    }
    _Float16* vb = Vm + (size_t)hb * DK_ * L_;
#pragma unroll
    for (int r = 0; r < 8; r++) vb[(m0 + r) * L_ + l] = (_Float16)av[r];
}

// ---------------------------------------------------------------------------
// Attention core (softmax over QUERY axis -> transposed formulation):
//   s'[j,i] = sum_dk kT[j,dk] * q[dk,i] / 4 ; p' = softmax_i per row j;
//   headT[j,v] = sum_i p'[j,i] * v[v,i]
// One WG per (hb, 16-row j chunk). LDS: 16x512 f32 scores + 16x512 f16 p.
// ---------------------------------------------------------------------------
__global__ void __launch_bounds__(128) k_attn(
        const _Float16* __restrict__ qT, const _Float16* __restrict__ kT,
        const _Float16* __restrict__ Vm, _Float16* __restrict__ headT) {
    __shared__ alignas(32) float    sbuf[16][L_];
    __shared__ alignas(32) _Float16 pbuf[16][L_];
    int wave = threadIdx.x >> 5, lane = threadIdx.x & 31;
    int j0 = blockIdx.x * 16;
    int hb = blockIdx.y;
    int b  = hb % B_, h = hb / B_;

    // Phase 1: s' tiles. A tile (16 j-rows) shared by all 32 N tiles.
    const _Float16* Aj = kT + ((size_t)hb * L_ + j0) * 32;
    const _Float16* Bq = qT + (size_t)hb * L_ * 32;
    v16h a = load_a16(Aj, 32);
    int nl = lane & 15, m0 = (lane < 16) ? 0 : 8;
#pragma unroll
    for (int it = 0; it < 8; it++) {
        int nt = wave * 8 + it;
        v16h bt = load_b16(Bq + nt * 16 * 32, 32);
        v8f acc = {};
        acc = wmma_f16(a, bt, acc);
#pragma unroll
        for (int r = 0; r < 8; r++) sbuf[m0 + r][nt * 16 + nl] = acc[r] * 0.25f;
    }
    __syncthreads();

    // Phase 2: row softmax (4 rows per wave).
#pragma unroll
    for (int rr = 0; rr < 4; rr++) {
        int row = wave * 4 + rr;
        float v[16];
        float mx = -1e30f;
#pragma unroll
        for (int i = 0; i < 16; i++) { v[i] = sbuf[row][lane + 32 * i]; mx = fmaxf(mx, v[i]); }
#pragma unroll
        for (int o = 16; o > 0; o >>= 1) mx = fmaxf(mx, __shfl_xor(mx, o));
        float s = 0.f;
#pragma unroll
        for (int i = 0; i < 16; i++) { v[i] = __expf(v[i] - mx); s += v[i]; }
#pragma unroll
        for (int o = 16; o > 0; o >>= 1) s += __shfl_xor(s, o);
        float inv = 1.f / s;
#pragma unroll
        for (int i = 0; i < 16; i++) pbuf[row][lane + 32 * i] = (_Float16)(v[i] * inv);
    }
    __syncthreads();

    // Phase 3: head tile = p'(16x512) @ v^T(512x16); K = 512 -> 16 WMMAs.
    if (wave == 0) {
        const _Float16* Vb = Vm + (size_t)hb * DK_ * L_;
        v8f acc = {};
        for (int kt = 0; kt < 16; kt++) {
            v16h pa = load_a16(&pbuf[0][kt * 32], L_);
            v16h vb = load_b16(Vb + kt * 32, L_);
            acc = wmma_f16(pa, vb, acc);
        }
#pragma unroll
        for (int r = 0; r < 8; r++)
            headT[((size_t)b * L_ + j0 + m0 + r) * C_ + h * DK_ + nl] = (_Float16)acc[r];
    }
}

// ---------------------------------------------------------------------------
extern "C" void kernel_launch(void* const* d_in, const int* in_sizes, int n_in,
                              void* d_out, int out_size, void* d_ws, size_t ws_size,
                              hipStream_t stream) {
    (void)in_sizes; (void)n_in; (void)out_size; (void)ws_size;
    const float* x    = (const float*)d_in[0];
    const float* dw_w = (const float*)d_in[1];
    const float* dw_b = (const float*)d_in[2];
    const float* pw_w = (const float*)d_in[3];
    const float* pw_b = (const float*)d_in[4];
    const float* Wq   = (const float*)d_in[5];
    const float* Wk   = (const float*)d_in[6];
    const float* Wv   = (const float*)d_in[7];
    const float* Wo   = (const float*)d_in[8];
    const float* Wfin = (const float*)d_in[9];
    float* out = (float*)d_out;

    const size_t BCL = (size_t)B_ * C_ * L_;         // 1,572,864
    char* w = (char*)d_ws;
    float*    cur   = (float*)w;              w += BCL * 4;                       // 6 MB
    float*    nrm   = (float*)w;              w += BCL * 4;                       // 6 MB
    _Float16* dwT   = (_Float16*)w;           w += BCL * 2;                       // 3 MB (B,L,C)
    _Float16* xnT   = (_Float16*)w;           w += BCL * 2;                       // 3 MB (B,L,C)
    _Float16* qT    = (_Float16*)w;           w += (size_t)H_ * B_ * L_ * 32 * 2; // 6 MB
    _Float16* kT    = (_Float16*)w;           w += (size_t)H_ * B_ * L_ * 32 * 2; // 6 MB
    _Float16* Vm    = (_Float16*)w;           w += (size_t)H_ * B_ * DK_ * L_ * 2;// 3 MB
    _Float16* headT = (_Float16*)w;           w += BCL * 2;                       // 3 MB (B,L,C)
    _Float16* pw16  = (_Float16*)w;           w += (size_t)NCONV_ * C_ * C_ * 2;
    _Float16* wq16  = (_Float16*)w;           w += (size_t)H_ * B_ * DK_ * C_ * 2;
    _Float16* wk16  = (_Float16*)w;           w += (size_t)H_ * B_ * DK_ * C_ * 2;
    _Float16* wv16  = (_Float16*)w;           w += (size_t)H_ * B_ * DK_ * C_ * 2;
    _Float16* wo16  = (_Float16*)w;           w += (size_t)B_ * C_ * C_ * 2;
    _Float16* wf16  = (_Float16*)w;           w += (size_t)B_ * C_ * C_ * 2;

    // f16 weight staging
    int nw;
    nw = NCONV_ * C_ * C_;      k_cvt<<<(nw + 255) / 256, 256, 0, stream>>>(pw_w, pw16, nw);
    nw = H_ * B_ * DK_ * C_;    k_cvt<<<(nw + 255) / 256, 256, 0, stream>>>(Wq,  wq16, nw);
                                k_cvt<<<(nw + 255) / 256, 256, 0, stream>>>(Wk,  wk16, nw);
                                k_cvt<<<(nw + 255) / 256, 256, 0, stream>>>(Wv,  wv16, nw);
    nw = B_ * C_ * C_;          k_cvt<<<(nw + 255) / 256, 256, 0, stream>>>(Wo,  wo16, nw);
                                k_cvt<<<(nw + 255) / 256, 256, 0, stream>>>(Wfin, wf16, nw);

    // positional encoding
    k_posenc<<<(unsigned)(BCL / 256), 256, 0, stream>>>(x, cur);

    // conv stack
    for (int ci = 0; ci < NCONV_; ci++) {
        k_norm<<<(B_ * C_) / 8, 256, 0, stream>>>(cur, nrm, nullptr);
        k_dwconv<<<(unsigned)(BCL / 256), 256, 0, stream>>>(nrm, dw_w, dw_b, ci, dwT);
        k_gemm<<<dim3(4, 8, B_), 256, 0, stream>>>(pw16 + (size_t)ci * C_ * C_, 0, dwT,
                                                   pw_b + ci * C_, cur, cur,
                                                   /*relu=*/1, /*seed=*/(unsigned)ci);
    }

    // attention
    k_norm<<<(B_ * C_) / 8, 256, 0, stream>>>(cur, nullptr, xnT);
    k_qkv<<<dim3(4, H_ * B_), 256, 0, stream>>>(wq16, wk16, wv16, xnT, qT, kT, Vm);
    k_attn<<<dim3(L_ / 16, H_ * B_), 128, 0, stream>>>(qT, kT, Vm, headT);
    k_gemm<<<dim3(4, 8, B_), 256, 0, stream>>>(wo16, C_ * C_, headT, nullptr, cur, cur,
                                               /*relu=*/0, /*seed=*/(unsigned)NCONV_);

    // final projection
    k_norm<<<(B_ * C_) / 8, 256, 0, stream>>>(cur, nullptr, xnT);
    k_gemm<<<dim3(4, 8, B_), 256, 0, stream>>>(wf16, C_ * C_, xnT, nullptr, cur, out,
                                               /*relu=*/1, /*seed=*/(unsigned)(NCONV_ + 1));
}
